// SphericalKMeans_24859270709684
// MI455X (gfx1250) — matrix operand order repaired and checked by reading, not compiled, not run on hardware
//
#include <hip/hip_runtime.h>

// ---------------------------------------------------------------------------
// Spherical K-Means assignment for MI455X (gfx1250, wave32, WMMA).
//   assignments[n] = argmax_k dot(vectors[n,:], centroids[k,:])
// Vector row-normalization is a positive per-row scale -> argmax invariant,
// so it is skipped; centroids are pre-normalized at init.
//
// Block = 256 thr (8 wave32), owns 128 rows in LDS (bf16, XOR-swizzled,
// 256 KB) and sweeps all 2048 centroids in 16-wide tiles, double-buffered
// through 2 x 32 KB LDS tiles staged with global_load_async_to_lds_b128
// (ASYNCcnt; zero staging VGPRs). Total LDS = 320 KB = full WGP LDS.
// Inner loop: 2-step-lookahead fragment pipeline + dual accumulators, with
// swizzle-periodic base pointers so LDS addresses fold into ds immediates.
// ---------------------------------------------------------------------------

typedef __attribute__((ext_vector_type(16))) __bf16 v16bf;
typedef __attribute__((ext_vector_type(8)))  __bf16 v8bf;
typedef __attribute__((ext_vector_type(8)))  float  v8f;

#define D_DIM      1024
#define K_CENT     2048
#define ROWS       128
#define KT_TILES   (K_CENT / 16)        // 128
#define ROWB       (D_DIM * 2)          // 2048 B per bf16 row (XOR swizzle, no pad)
#define A_BYTES    (ROWS * ROWB)        // 262144
#define BTILE_B    (16 * ROWB)          // 32768
#define SMEM_BYTES (A_BYTES + 2 * BTILE_B)  // 327680 = 320 KB exactly

static_assert(SMEM_BYTES == 320u * 1024u, "expect exact WGP LDS fit");

struct __align__(8) bf16x4 { __bf16 x, y, z, w; };

// round-to-nearest-even f32 -> bf16
__device__ __forceinline__ __bf16 f2bf(float f) {
  unsigned u = __builtin_bit_cast(unsigned, f);
  u += 0x7FFFu + ((u >> 16) & 1u);
  unsigned short s = (unsigned short)(u >> 16);
  return __builtin_bit_cast(__bf16, s);
}

__device__ __forceinline__ v16bf cat16(v8bf lo, v8bf hi) {
  return __builtin_shufflevector(lo, hi, 0, 1, 2, 3, 4, 5, 6, 7,
                                 8, 9, 10, 11, 12, 13, 14, 15);
}

// CDNA5 async copy: global -> LDS, 16B per lane, tracked by ASYNCcnt.
__device__ __forceinline__ void async_ld_b128(unsigned lds_off, const void* gaddr) {
  asm volatile("global_load_async_to_lds_b128 %0, %1, off"
               :: "v"(lds_off), "v"(gaddr) : "memory");
}
__device__ __forceinline__ void wait_asynccnt0() {
  asm volatile("s_wait_asynccnt 0x0" ::: "memory");
}

// ---------------------------------------------------------------------------
// Kernel 0: centroids f32 [K,D] -> bf16 [K,D] workspace (4 MB, L2-resident)
// ---------------------------------------------------------------------------
__global__ void __launch_bounds__(256)
cvt_centroids_bf16(const float* __restrict__ cin, __bf16* __restrict__ cout, int n4) {
  int i = blockIdx.x * blockDim.x + threadIdx.x;
  if (i < n4) {
    float4 f = reinterpret_cast<const float4*>(cin)[i];
    bf16x4 o = {f2bf(f.x), f2bf(f.y), f2bf(f.z), f2bf(f.w)};
    reinterpret_cast<bf16x4*>(cout)[i] = o;
  }
}

// ---------------------------------------------------------------------------
// Kernel 1: fused WMMA GEMM + running argmax
// ---------------------------------------------------------------------------
__global__ void __launch_bounds__(256)
spherical_assign_kernel(const float* __restrict__ vectors,
                        const __bf16* __restrict__ cent,
                        int* __restrict__ assignments) {
  extern __shared__ char smem[];

  const int tid  = threadIdx.x;
  const int lane = tid & 31;
  const int wv   = tid >> 5;      // wave 0..7
  const int hi   = lane >> 4;     // fragment K-half per ISA layout
  const int ln   = lane & 15;

  const int row0 = blockIdx.x * ROWS;

  // ---- Phase 1: 128 rows f32 -> bf16 into LDS, XOR-swizzled 16B chunks ----
  {
    const int c  = tid >> 1;      // chunk 0..127 of a row
    const int hf = tid & 1;       // 8B half within the chunk
#pragma unroll 4
    for (int r = 0; r < ROWS; ++r) {
      float4 f = reinterpret_cast<const float4*>(
          vectors + (size_t)(row0 + r) * D_DIM)[tid];
      bf16x4 o = {f2bf(f.x), f2bf(f.y), f2bf(f.z), f2bf(f.w)};
      *reinterpret_cast<bf16x4*>(
          smem + r * ROWB + (((c ^ (r & 15)) << 4) | (hf << 3))) = o;
    }
  }

  // ---- B-tile staging (async, 32 KB tile: 16 centroids x 1024 bf16) ----
  const int sc = tid >> 4;            // centroid 0..15 staged by this thread
  const int sj = (tid & 15) * 8;      // first of 8 16B-chunks staged
  const unsigned smemBase = (unsigned)(size_t)smem;   // LDS byte offset of smem

  auto stage_tile_async = [&](int kt_i, int buf) {
    const __bf16* srow = cent + (size_t)(kt_i * 16 + sc) * D_DIM;
    const unsigned drow = smemBase + A_BYTES + buf * BTILE_B + sc * ROWB;
#pragma unroll
    for (int i = 0; i < 8; ++i) {
      const int j = sj + i;
      async_ld_b128(drow + (((unsigned)(j ^ sc)) << 4), srow + j * 8);
    }
  };

  // initial stage: tile 0 -> buffer 0 (overlaps the phase-1 LDS fill)
  stage_tile_async(0, 0);
  wait_asynccnt0();
  __syncthreads();

  // ---- swizzle-periodic fragment base pointers ----
  // A chunk index for step s:  lo: 4s+hi, hi-half: 4s+hi+2 -> low 4 bits have
  // period 4 in s; high bits advance +256B linearly => 4 bases per stream,
  // all other steps addressed via ds immediate offsets (m*256).
  const char* aRow = smem + (size_t)(wv * 16 + ln) * ROWB;   // swizzle key == ln
  const char* aPtrLo[4], *aPtrHi[4];
  unsigned    bOffLo[4], bOffHi[4];
#pragma unroll
  for (int j = 0; j < 4; ++j) {
    aPtrLo[j] = aRow + ((unsigned)((4 * j + hi) ^ ln) << 4);
    aPtrHi[j] = aRow + ((unsigned)((4 * j + hi + 2) ^ ln) << 4);
    bOffLo[j] = (unsigned)(ln * ROWB) + ((unsigned)((4 * j + hi) ^ ln) << 4);
    bOffHi[j] = (unsigned)(ln * ROWB) + ((unsigned)((4 * j + hi + 2) ^ ln) << 4);
  }

  float best[8];
  int   bidx[8];
#pragma unroll
  for (int r = 0; r < 8; ++r) { best[r] = -3.402823466e38f; bidx[r] = 0; }

  for (int k = 0; k < KT_TILES; ++k) {
    const int p = k & 1;

    // ---- kick off next tile's async stage (overlaps the WMMA block) ----
    if (k + 1 < KT_TILES) {
      stage_tile_async(k + 1, p ^ 1);
      if (k + 2 < KT_TILES)   // global_prefetch_b8 two tiles ahead
        __builtin_prefetch(cent + (size_t)((k + 2) * 16 + sc) * D_DIM + sj * 8, 0, 1);
    }

    // per-tile B base pointers (one v_add per base, immediates thereafter)
    const char* bBuf = smem + A_BYTES + (size_t)p * BTILE_B;
    const char* bPtrLo[4], *bPtrHi[4];
#pragma unroll
    for (int j = 0; j < 4; ++j) {
      bPtrLo[j] = bBuf + bOffLo[j];
      bPtrHi[j] = bBuf + bOffHi[j];
    }

    auto loadA = [&](int s) -> v16bf {
      const int j = s & 3, m = s >> 2;
      v8bf lo = *reinterpret_cast<const v8bf*>(aPtrLo[j] + (m << 8));
      v8bf h8 = *reinterpret_cast<const v8bf*>(aPtrHi[j] + (m << 8));
      return cat16(lo, h8);
    };
    auto loadB = [&](int s) -> v16bf {
      const int j = s & 3, m = s >> 2;
      v8bf lo = *reinterpret_cast<const v8bf*>(bPtrLo[j] + (m << 8));
      v8bf h8 = *reinterpret_cast<const v8bf*>(bPtrHi[j] + (m << 8));
      return cat16(lo, h8);
    };

    // ---- 32 K-steps, 2-step-lookahead pipeline, dual accumulators ----
    v8f acc0 = {}, acc1 = {};
    v16bf a0 = loadA(0), b0 = loadB(0);
    v16bf a1 = loadA(1), b1 = loadB(1);
#pragma unroll
    for (int s = 0; s < 32; ++s) {
      v16bf an = a0, bn = b0;
      if (s + 2 < 32) { an = loadA(s + 2); bn = loadB(s + 2); }
      if (s & 1)
        acc1 = __builtin_amdgcn_wmma_f32_16x16x32_bf16(
            false, a0, false, b0, (short)0, acc1, false, false);
      else
        acc0 = __builtin_amdgcn_wmma_f32_16x16x32_bf16(
            false, a0, false, b0, (short)0, acc0, false, false);
      a0 = a1; b0 = b1; a1 = an; b1 = bn;
    }

    // ---- fold tile into running argmax (col = k*16 + ln) ----
    const int cb = k * 16 + ln;
#pragma unroll
    for (int r = 0; r < 8; ++r) {
      float v = acc0[r] + acc1[r];
      if (v > best[r]) { best[r] = v; bidx[r] = cb; }   // strict '>' keeps lowest k
    }

    // ---- async stage complete -> visible after barrier ----
    if (k + 1 < KT_TILES) wait_asynccnt0();
    __syncthreads();
  }

  // ---- final argmax across the 16 lanes sharing each row (first-max wins) ----
#pragma unroll
  for (int r = 0; r < 8; ++r) {
    float v  = best[r];
    int   id = bidx[r];
#pragma unroll
    for (int off = 8; off >= 1; off >>= 1) {
      float v2 = __shfl_xor(v, off, 16);
      int   i2 = __shfl_xor(id, off, 16);
      if (v2 > v || (v2 == v && i2 < id)) { v = v2; id = i2; }
    }
    if (ln == 0)
      assignments[row0 + wv * 16 + hi * 8 + r] = id;
  }
}

// ---------------------------------------------------------------------------
// Host launcher (graph-capture safe)
// ---------------------------------------------------------------------------
extern "C" void kernel_launch(void* const* d_in, const int* in_sizes, int n_in,
                              void* d_out, int out_size, void* d_ws, size_t ws_size,
                              hipStream_t stream) {
  const float* vectors   = (const float*)d_in[0];   // [N, D] f32
  const float* centroids = (const float*)d_in[1];   // [K, D] f32 (pre-normalized)
  int* assignments = (int*)d_out;                   // [N] int32

  __bf16* cent_bf16 = (__bf16*)d_ws;                // K*D*2 = 4 MB scratch

  const int N = in_sizes[0] / D_DIM;                // 131072

  const int n4 = K_CENT * D_DIM / 4;
  cvt_centroids_bf16<<<(n4 + 255) / 256, 256, 0, stream>>>(centroids, cent_bf16, n4);

  hipFuncSetAttribute(reinterpret_cast<const void*>(&spherical_assign_kernel),
                      hipFuncAttributeMaxDynamicSharedMemorySize, (int)SMEM_BYTES);
  spherical_assign_kernel<<<N / ROWS, 256, SMEM_BYTES, stream>>>(
      vectors, cent_bf16, assignments);
}
